// VoxelBackBone8x_69269232550635
// MI455X (gfx1250) — matrix operand order repaired and checked by reading, not compiled, not run on hardware
//
#include <hip/hip_runtime.h>

typedef __attribute__((ext_vector_type(16))) __bf16 v16bf;
typedef __attribute__((ext_vector_type(8)))  float  v8f;

#define EPS_ 0.001f

__device__ __forceinline__ unsigned short f2bf(float f) {
    unsigned int u = __float_as_uint(f);
    unsigned int r = u + 0x7FFFu + ((u >> 16) & 1u);
    return (unsigned short)(r >> 16);
}

// ---------------- utility kernels ----------------
__global__ void k_zero(float* p, long n) {
    long i = (long)blockIdx.x * blockDim.x + threadIdx.x;
    long st = (long)gridDim.x * blockDim.x;
    for (; i < n; i += st) p[i] = 0.0f;
}

__global__ void k_scatter(const float* __restrict__ feats, const int* __restrict__ coords,
                          float* __restrict__ dense, float* __restrict__ mask, int N) {
    int i = blockIdx.x * blockDim.x + threadIdx.x;
    if (i >= N) return;
    int z = coords[i * 4 + 1], y = coords[i * 4 + 2], x = coords[i * 4 + 3];
    long cell = ((long)z * 160 + y) * 160 + x;
    atomicAdd(&dense[cell * 4 + 0], feats[i * 4 + 0]);
    atomicAdd(&dense[cell * 4 + 1], feats[i * 4 + 1]);
    atomicAdd(&dense[cell * 4 + 2], feats[i * 4 + 2]);
    atomicAdd(&dense[cell * 4 + 3], feats[i * 4 + 3]);
    mask[cell] = 1.0f;
}

__global__ void k_f32_to_bf16(const float* __restrict__ in, unsigned short* __restrict__ out, long n) {
    long i = (long)blockIdx.x * blockDim.x + threadIdx.x;
    if (i < n) out[i] = f2bf(in[i]);
}

// weights (kz,ky,kx,Cin,Cout) f32  ->  [tap][Cout][KP] bf16, zero-padded K >= Cin
__global__ void k_wtrans(const float* __restrict__ w, unsigned short* __restrict__ wT,
                         int taps, int Cin, int Cout, int KP) {
    long idx = (long)blockIdx.x * blockDim.x + threadIdx.x;
    long total = (long)taps * Cout * KP;
    if (idx >= total) return;
    int k = (int)(idx % KP);
    long r = idx / KP;
    int co = (int)(r % Cout);
    int t  = (int)(r / Cout);
    float val = (k < Cin) ? w[((long)t * Cin + k) * Cout + co] : 0.0f;
    wT[idx] = f2bf(val);
}

// tap-pair layout for Cin<=16: [pair][Cout][32], K0-15 = tap 2p, K16-31 = tap 2p+1
__global__ void k_wtrans_pair(const float* __restrict__ w, unsigned short* __restrict__ wT,
                              int taps, int Cin, int Cout) {
    long idx = (long)blockIdx.x * blockDim.x + threadIdx.x;
    int np = (taps + 1) >> 1;
    long total = (long)np * Cout * 32;
    if (idx >= total) return;
    int k = (int)(idx % 32);
    long r = idx / 32;
    int co = (int)(r % Cout);
    int tp = (int)(r / Cout);
    int t  = tp * 2 + (k >> 4);
    int ci = k & 15;
    float val = (t < taps && ci < Cin) ? w[((long)t * Cin + ci) * Cout + co] : 0.0f;
    wT[idx] = f2bf(val);
}

// ---------------- WMMA implicit-GEMM 3D conv (tap loop) ----------------
// one wave per 16(M) x 16*NT(N) output tile; A rows gathered per lane,
// matching the CDNA5 16-bit A (16x32) / B (32x16) VGPR layouts exactly.
// PAIR=1: two taps packed per K=32 WMMA (for Cin <= 16).
template<int NT, int PAIR>
__global__ __launch_bounds__(32) void k_conv_wmma(
    const unsigned short* __restrict__ in, const unsigned short* __restrict__ wT,
    float* __restrict__ out, const unsigned short* __restrict__ zp,
    int Din, int Hin, int Win, int Cin,
    int Dout, int Hout, int Wout, int Cout,
    int kz, int ky, int kx, int sz, int sy, int sx, int pz, int py, int px, int KP)
{
    const int lane = threadIdx.x;
    const int hi   = lane >> 4;          // 0: lanes 0-15, 1: lanes 16-31
    const int klo  = hi ? 8 : 0;         // A chunk-local K: klo..klo+7, klo+16..klo+23
    const int kbB  = hi ? 8 : 0;         // B chunk-local K: kbB..kbB+7, kbB+16..kbB+23
    const int ncol = lane & 15;
    const int M    = Dout * Hout * Wout;
    const int m0   = blockIdx.x * 16;
    const int nb   = blockIdx.y * (NT * 16);

    int  vm   = m0 + (lane & 15);
    bool mrow = vm < M;
    int  vmc  = mrow ? vm : 0;
    int  od   = vmc / (Hout * Wout);
    int  r2   = vmc - od * (Hout * Wout);
    int  oh   = r2 / Wout;
    int  ow   = r2 - oh * Wout;

    v8f acc[NT];
#pragma unroll
    for (int nt = 0; nt < NT; ++nt)
        acc[nt] = (v8f){0.f, 0.f, 0.f, 0.f, 0.f, 0.f, 0.f, 0.f};

    const int ntap = kz * ky * kx;

    if (PAIR) {
        const int np = (ntap + 1) >> 1;
        for (int tp = 0; tp < np; ++tp) {
            const unsigned short* rowp[2];
#pragma unroll
            for (int j = 0; j < 2; ++j) {
                int t  = tp * 2 + j;
                int tz = t / (ky * kx);
                int tr = t - tz * (ky * kx);
                int ty = tr / kx;
                int tx = tr - ty * kx;
                int id = od * sz + tz - pz;
                int ih = oh * sy + ty - py;
                int iw = ow * sx + tx - px;
                bool ok = (t < ntap) & mrow & ((unsigned)id < (unsigned)Din) &
                          ((unsigned)ih < (unsigned)Hin) & ((unsigned)iw < (unsigned)Win);
                size_t off = ok ? (size_t)((((long)id * Hin + ih) * Win + iw)) * (size_t)Cin : 0;
                rowp[j] = ok ? (in + off) : zp;
            }
            union { uint2 u[4]; v16bf v; } A;
            const char* rp0 = (const char*)(rowp[0] + klo);
            const char* rp1 = (const char*)(rowp[1] + klo);
            A.u[0] = *(const uint2*)(rp0 + 0);
            A.u[1] = *(const uint2*)(rp0 + 8);
            A.u[2] = *(const uint2*)(rp1 + 0);   // K16.. = second tap
            A.u[3] = *(const uint2*)(rp1 + 8);
#pragma unroll
            for (int nt = 0; nt < NT; ++nt) {
                int co = nb + nt * 16 + ncol;
                union { uint4 u[2]; v16bf v; } B;
                const uint4* bp = (const uint4*)(wT + ((size_t)tp * Cout + co) * 32 + kbB);
                B.u[0] = bp[0];
                B.u[1] = bp[2];
                acc[nt] = __builtin_amdgcn_wmma_f32_16x16x32_bf16(
                    false, A.v, false, B.v, (short)0, acc[nt], false, false);
            }
        }
    } else {
        const int KC = KP >> 5;
        for (int t = 0; t < ntap; ++t) {
            int tz = t / (ky * kx);
            int tr = t - tz * (ky * kx);
            int ty = tr / kx;
            int tx = tr - ty * kx;
            int id = od * sz + tz - pz;
            int ih = oh * sy + ty - py;
            int iw = ow * sx + tx - px;
            bool ok = mrow & ((unsigned)id < (unsigned)Din) &
                             ((unsigned)ih < (unsigned)Hin) &
                             ((unsigned)iw < (unsigned)Win);
            size_t off = ok ? (size_t)((((long)id * Hin + ih) * Win + iw)) * (size_t)Cin : 0;
            const unsigned short* rowp = ok ? (in + off) : zp;

            for (int kc = 0; kc < KC; ++kc) {
                union { uint2 u[4]; v16bf v; } A;
                const char* rp = (const char*)(rowp + kc * 32 + klo);
                A.u[0] = *(const uint2*)(rp + 0);
                A.u[1] = *(const uint2*)(rp + 8);
                A.u[2] = *(const uint2*)(rp + 32);   // K + 16
                A.u[3] = *(const uint2*)(rp + 40);
#pragma unroll
                for (int nt = 0; nt < NT; ++nt) {
                    int co = nb + nt * 16 + ncol;
                    union { uint4 u[2]; v16bf v; } B;
                    const uint4* bp = (const uint4*)(wT + ((size_t)t * Cout + co) * KP + kc * 32 + kbB);
                    B.u[0] = bp[0];
                    B.u[1] = bp[2];
                    acc[nt] = __builtin_amdgcn_wmma_f32_16x16x32_bf16(
                        false, A.v, false, B.v, (short)0, acc[nt], false, false);
                }
            }
        }
    }

#pragma unroll
    for (int nt = 0; nt < NT; ++nt) {
#pragma unroll
        for (int r = 0; r < 8; ++r) {
            int m = r + hi * 8;
            int v = m0 + m;
            if (v < M) out[(size_t)v * Cout + nb + nt * 16 + ncol] = acc[nt][r];
        }
    }
}

// ---------------- masked BN statistics ----------------
// S[0..C-1] = sum(x*mask), S[128..128+C-1] = sum((x*mask)^2), S[256] = sum(mask)
__global__ void k_bn_stats(const float* __restrict__ x, const float* __restrict__ mask,
                           float* __restrict__ S, int V, int C, int log2C)
{
    __shared__ float s1[128];
    __shared__ float s2[128];
    __shared__ float sc;
    int t = threadIdx.x;
    for (int i = t; i < C; i += blockDim.x) { s1[i] = 0.f; s2[i] = 0.f; }
    if (t == 0) sc = 0.f;
    __syncthreads();
    int c = t & (C - 1);
    int vg = t >> log2C;
    int groups = blockDim.x >> log2C;
    float a1 = 0.f, a2 = 0.f, am = 0.f;
    for (int v = blockIdx.x * groups + vg; v < V; v += gridDim.x * groups) {
        float m = mask[v];
        float val = x[(size_t)v * C + c] * m;
        a1 += val; a2 += val * val;
        if (c == 0) am += m;
    }
    atomicAdd(&s1[c], a1);
    atomicAdd(&s2[c], a2);
    if (c == 0) atomicAdd(&sc, am);
    __syncthreads();
    for (int i = t; i < C; i += blockDim.x) {
        atomicAdd(&S[i], s1[i]);
        atomicAdd(&S[128 + i], s2[i]);
    }
    if (t == 0) atomicAdd(&S[256], sc);
}

template<int BF16OUT>
__global__ void k_bn_apply(const float* __restrict__ x, const float* __restrict__ mask,
                           const float* __restrict__ S,
                           const float* __restrict__ g, const float* __restrict__ b,
                           void* __restrict__ out, int V, int C)
{
    long idx = (long)blockIdx.x * blockDim.x + threadIdx.x;
    long n = (long)V * C;
    if (idx >= n) return;
    int c = (int)(idx % C);
    int v = (int)(idx / C);
    float cnt  = fmaxf(S[256], 1.0f);
    float mean = S[c] / cnt;
    float var  = S[128 + c] / cnt - mean * mean;
    float y = g[c] * (x[idx] - mean) * rsqrtf(var + EPS_) + b[c];
    y = fmaxf(y, 0.0f) * mask[v];
    if (BF16OUT) ((unsigned short*)out)[idx] = f2bf(y);
    else         ((float*)out)[idx] = y;
}

// ---------------- mask dilation (conv with ones > 0) ----------------
__global__ void k_dilate(const float* __restrict__ mi, float* __restrict__ mo,
                         int Din, int Hin, int Win, int Dout, int Hout, int Wout,
                         int kz, int ky, int kx, int sz, int sy, int sx,
                         int pz, int py, int px)
{
    int idx = blockIdx.x * blockDim.x + threadIdx.x;
    int M = Dout * Hout * Wout;
    if (idx >= M) return;
    int od = idx / (Hout * Wout);
    int r  = idx - od * (Hout * Wout);
    int oh = r / Wout;
    int ow = r - oh * Wout;
    float any = 0.f;
    for (int tz = 0; tz < kz; ++tz)
        for (int ty = 0; ty < ky; ++ty)
            for (int tx = 0; tx < kx; ++tx) {
                int id = od * sz + tz - pz;
                int ih = oh * sy + ty - py;
                int iw = ow * sx + tx - px;
                if ((unsigned)id < (unsigned)Din && (unsigned)ih < (unsigned)Hin &&
                    (unsigned)iw < (unsigned)Win)
                    if (mi[((long)id * Hin + ih) * Win + iw] > 0.f) any = 1.f;
            }
    mo[idx] = any;
}

// ---------------- host side ----------------
struct ConvDesc {
    int taps, Cin, Cout, KP, NT, pair;
    int Din, Hin, Win, Dout, Hout, Wout;
    int kz, ky, kx, sz, sy, sx, pz, py, px;
};

static void launch_conv(hipStream_t s, const ConvDesc& d,
                        const unsigned short* in, const unsigned short* wT, float* out,
                        const unsigned short* zp)
{
    int M = d.Dout * d.Hout * d.Wout;
    dim3 grid((M + 15) / 16, d.Cout / (d.NT * 16));
    dim3 blk(32);
#define CONV_ARGS in, wT, out, zp, d.Din, d.Hin, d.Win, d.Cin, d.Dout, d.Hout, d.Wout, \
                  d.Cout, d.kz, d.ky, d.kx, d.sz, d.sy, d.sx, d.pz, d.py, d.px, d.KP
    if (d.pair) {
        if (d.NT == 1) k_conv_wmma<1, 1><<<grid, blk, 0, s>>>(CONV_ARGS);
        else           k_conv_wmma<2, 1><<<grid, blk, 0, s>>>(CONV_ARGS);
    } else {
        switch (d.NT) {
            case 1:  k_conv_wmma<1, 0><<<grid, blk, 0, s>>>(CONV_ARGS); break;
            case 2:  k_conv_wmma<2, 0><<<grid, blk, 0, s>>>(CONV_ARGS); break;
            default: k_conv_wmma<4, 0><<<grid, blk, 0, s>>>(CONV_ARGS); break;
        }
    }
#undef CONV_ARGS
}

extern "C" void kernel_launch(void* const* d_in, const int* in_sizes, int n_in,
                              void* d_out, int out_size, void* d_ws, size_t ws_size,
                              hipStream_t stream)
{
    (void)in_sizes; (void)n_in; (void)out_size;
    // 12 conv blocks in setup_inputs insertion order:
    // conv_input, conv1, conv2[0..2], conv3[0..2], conv4[0..2], conv_out; each (w,g,b)
    const float* W[12]; const float* G[12]; const float* Bp[12];
    for (int i = 0; i < 12; ++i) {
        W[i]  = (const float*)d_in[3 * i + 0];
        G[i]  = (const float*)d_in[3 * i + 1];
        Bp[i] = (const float*)d_in[3 * i + 2];
    }
    const float* feats  = (const float*)d_in[36];
    const int*   coords = (const int*)d_in[37];
    const int N = 40000;

    const long V0 = 41L * 160 * 160;   // 1,049,600
    const long V2 = 21L * 80 * 80;     //   134,400
    const long V3 = 11L * 40 * 40;     //    17,600
    const long V4 = 5L * 20 * 20;      //     2,000
    const long VO = 2L * 20 * 20;      //       800

    char* ws = (char*)d_ws;
    size_t cur = 0;
    auto alloc = [&](size_t bytes) { size_t o = cur; cur = (cur + bytes + 255) & ~(size_t)255; return o; };

    size_t off_zp     = alloc(256);
    size_t off_dense0 = alloc((size_t)V0 * 4 * 4);
    size_t off_mask0  = alloc((size_t)V0 * 4);
    size_t zero_bytes = cur;                       // zero [0, cur) each launch
    size_t off_bufA   = alloc((size_t)V0 * 16 * 2 + 256);
    size_t off_bufB   = alloc((size_t)V0 * 16 * 2 + 256);
    size_t off_F      = alloc((size_t)V0 * 16 * 4);
    size_t off_m2     = alloc((size_t)V2 * 4);
    size_t off_m3     = alloc((size_t)V3 * 4);
    size_t off_m4     = alloc((size_t)V4 * 4);
    size_t off_mo     = alloc((size_t)VO * 4);
    size_t off_S      = alloc(257 * 4);

    static const ConvDesc cd[12] = {
        // taps Cin Cout KP NT pr  Din Hin Win  Dout Hout Wout  k       s       p
        {27,  4,  16, 32, 1, 1,  41,160,160, 41,160,160, 3,3,3, 1,1,1, 1,1,1},  // conv_input
        {27, 16,  16, 32, 1, 1,  41,160,160, 41,160,160, 3,3,3, 1,1,1, 1,1,1},  // conv1
        {27, 16,  32, 32, 2, 1,  41,160,160, 21, 80, 80, 3,3,3, 2,2,2, 1,1,1},  // conv2[0]
        {27, 32,  32, 32, 2, 0,  21, 80, 80, 21, 80, 80, 3,3,3, 1,1,1, 1,1,1},  // conv2[1]
        {27, 32,  32, 32, 2, 0,  21, 80, 80, 21, 80, 80, 3,3,3, 1,1,1, 1,1,1},  // conv2[2]
        {27, 32,  64, 32, 4, 0,  21, 80, 80, 11, 40, 40, 3,3,3, 2,2,2, 1,1,1},  // conv3[0]
        {27, 64,  64, 64, 4, 0,  11, 40, 40, 11, 40, 40, 3,3,3, 1,1,1, 1,1,1},  // conv3[1]
        {27, 64,  64, 64, 4, 0,  11, 40, 40, 11, 40, 40, 3,3,3, 1,1,1, 1,1,1},  // conv3[2]
        {27, 64,  64, 64, 4, 0,  11, 40, 40,  5, 20, 20, 3,3,3, 2,2,2, 0,1,1},  // conv4[0]
        {27, 64,  64, 64, 4, 0,   5, 20, 20,  5, 20, 20, 3,3,3, 1,1,1, 1,1,1},  // conv4[1]
        {27, 64,  64, 64, 4, 0,   5, 20, 20,  5, 20, 20, 3,3,3, 1,1,1, 1,1,1},  // conv4[2]
        { 3, 64, 128, 64, 4, 0,   5, 20, 20,  2, 20, 20, 3,1,1, 2,1,1, 0,0,0},  // conv_out
    };

    size_t off_wT[12];
    for (int i = 0; i < 12; ++i) {
        size_t wbytes = cd[i].pair
            ? (size_t)((cd[i].taps + 1) / 2) * cd[i].Cout * 32 * 2
            : (size_t)cd[i].taps * cd[i].Cout * cd[i].KP * 2;
        off_wT[i] = alloc(wbytes);
    }

    if (ws_size < cur) return;  // insufficient workspace; nothing safe to do

    unsigned short* zp    = (unsigned short*)(ws + off_zp);
    float*          dense0= (float*)(ws + off_dense0);
    float*          mask0 = (float*)(ws + off_mask0);
    unsigned short* bufA  = (unsigned short*)(ws + off_bufA);
    unsigned short* bufB  = (unsigned short*)(ws + off_bufB);
    float*          F     = (float*)(ws + off_F);
    float*          S     = (float*)(ws + off_S);
    float* masks_all[5]   = { mask0, (float*)(ws + off_m2), (float*)(ws + off_m3),
                              (float*)(ws + off_m4), (float*)(ws + off_mo) };

    // 1) zero [zero-page | dense grid | mask grid]
    k_zero<<<4096, 256, 0, stream>>>((float*)ws, (long)(zero_bytes / 4));
    // 2) scatter voxels
    k_scatter<<<(N + 255) / 256, 256, 0, stream>>>(feats, coords, dense0, mask0, N);
    // 3) dense f32 -> bf16 input
    k_f32_to_bf16<<<(int)((V0 * 4 + 255) / 256), 256, 0, stream>>>(dense0, bufA, V0 * 4);
    // 4) weight transforms
    for (int i = 0; i < 12; ++i) {
        unsigned short* wTp = (unsigned short*)(ws + off_wT[i]);
        if (cd[i].pair) {
            long tot = (long)((cd[i].taps + 1) / 2) * cd[i].Cout * 32;
            k_wtrans_pair<<<(int)((tot + 255) / 256), 256, 0, stream>>>(
                W[i], wTp, cd[i].taps, cd[i].Cin, cd[i].Cout);
        } else {
            long tot = (long)cd[i].taps * cd[i].Cout * cd[i].KP;
            k_wtrans<<<(int)((tot + 255) / 256), 256, 0, stream>>>(
                W[i], wTp, cd[i].taps, cd[i].Cin, cd[i].Cout, cd[i].KP);
        }
    }

    // mask index used by each conv's BN, and which convs dilate a new mask
    static const int mask_of[12] = {0, 0, 1, 1, 1, 2, 2, 2, 3, 3, 3, 4};
    static const int dil_at[12]  = {0, 0, 1, 0, 0, 1, 0, 0, 1, 0, 0, 1};

    unsigned short* pin = bufA;
    unsigned short* pout = bufB;
    for (int i = 0; i < 12; ++i) {
        const ConvDesc& d = cd[i];
        int V = d.Dout * d.Hout * d.Wout;
        int C = d.Cout;

        if (dil_at[i]) {
            const float* min_ = masks_all[mask_of[i] - 1];
            float* mo_ = masks_all[mask_of[i]];
            k_dilate<<<(V + 255) / 256, 256, 0, stream>>>(
                min_, mo_, d.Din, d.Hin, d.Win, d.Dout, d.Hout, d.Wout,
                d.kz, d.ky, d.kx, d.sz, d.sy, d.sx, d.pz, d.py, d.px);
        }
        const float* mk = masks_all[mask_of[i]];

        launch_conv(stream, d, pin, (const unsigned short*)(ws + off_wT[i]), F, zp);

        k_zero<<<1, 256, 0, stream>>>(S, 257);
        int log2C = 31 - __builtin_clz((unsigned)C);
        k_bn_stats<<<256, 256, 0, stream>>>(F, mk, S, V, C, log2C);

        long n = (long)V * C;
        int blocks = (int)((n + 255) / 256);
        if (i == 11) {
            k_bn_apply<0><<<blocks, 256, 0, stream>>>(F, mk, S, G[i], Bp[i], d_out, V, C);
        } else {
            k_bn_apply<1><<<blocks, 256, 0, stream>>>(F, mk, S, G[i], Bp[i], (void*)pout, V, C);
            unsigned short* t = pin; pin = pout; pout = t;
        }
    }
}
